// ncodLoss_50397146251377
// MI455X (gfx1250) — matrix operand order repaired and checked by reading, not compiled, not run on hardware
//
#include <hip/hip_runtime.h>
#include <math.h>

typedef _Float16 v16h __attribute__((ext_vector_type(16)));
typedef _Float16 h8   __attribute__((ext_vector_type(8)));
typedef _Float16 h4   __attribute__((ext_vector_type(4)));
typedef float    v8f  __attribute__((ext_vector_type(8)));

#define N_EXAMP 500000
#define BATCH   16384
#define NCLS    1000
#define FEAT    512
#define EPSV    1e-4f

// Segment-sum tiling: 8 feature blocks of 64 feats, 32 example chunks.
#define SEG_FB 8
#define SEG_EC 32
#define SEG_CHUNK ((N_EXAMP + SEG_EC - 1) / SEG_EC)   // 15625

// ---------------------------------------------------------------- zero scratch
__global__ void zeroK(float* __restrict__ p, int n) {
  int i = blockIdx.x * blockDim.x + threadIdx.x;
  int stride = gridDim.x * blockDim.x;
  for (; i < n; i += stride) p[i] = 0.f;
}

// ---------------------------------------------------------------- class counts
__global__ void countK(const int* __restrict__ labels, float* __restrict__ counts) {
  __shared__ float lc[NCLS];
  const int t = threadIdx.x;
  for (int j = t; j < NCLS; j += blockDim.x) lc[j] = 0.f;
  __syncthreads();
  int per = (N_EXAMP + gridDim.x - 1) / gridDim.x;
  int e0 = blockIdx.x * per;
  int e1 = e0 + per; if (e1 > N_EXAMP) e1 = N_EXAMP;
  for (int e = e0 + t; e < e1; e += blockDim.x)
    atomicAdd(&lc[labels[e]], 1.f);
  __syncthreads();
  for (int j = t; j < NCLS; j += blockDim.x) {
    float v = lc[j];
    if (v != 0.f) atomicAdd(&counts[j], v);
  }
}

// --------------------------------------------- segment sum via 256KB LDS tiles
// block = (fb, ec): accumulate 64 feats of feature-block fb over example chunk
// ec into LDS[1000][64], then flush with global atomics.
__global__ void segsumK(const float* __restrict__ prevSim,
                        const int* __restrict__ labels,
                        float* __restrict__ mvsum) {
  extern __shared__ float acc[];                 // NCLS*64 floats = 256 KB
  const int fb = blockIdx.x % SEG_FB;
  const int ec = blockIdx.x / SEG_FB;
  const int t  = threadIdx.x;
  for (int j = t; j < NCLS * 64; j += blockDim.x) acc[j] = 0.f;
  __syncthreads();
  const int lane = t & 31, w = t >> 5;
  const int nw = blockDim.x >> 5;                // 8 waves
  const int e0 = ec * SEG_CHUNK;
  int e1 = e0 + SEG_CHUNK; if (e1 > N_EXAMP) e1 = N_EXAMP;
  for (int e = e0 + w; e < e1; e += nw) {
    if (e + nw < e1)
      __builtin_prefetch(prevSim + (size_t)(e + nw) * FEAT + fb * 64, 0, 0);
    const int cls = labels[e];
    const float2 v = ((const float2*)(prevSim + (size_t)e * FEAT + fb * 64))[lane];
    atomicAdd(&acc[cls * 64 + lane * 2    ], v.x);   // ds_add_f32, bank-conflict free
    atomicAdd(&acc[cls * 64 + lane * 2 + 1], v.y);
  }
  __syncthreads();
  for (int j = t; j < NCLS * 64; j += blockDim.x) {
    float v = acc[j];
    if (v != 0.f)
      atomicAdd(&mvsum[(size_t)(j >> 6) * FEAT + fb * 64 + (j & 63)], v);
  }
}

// -------------------------------------- masterVector mean + L2-normalize (f16)
__global__ void normMvK(const float* __restrict__ mvsum,
                        const float* __restrict__ counts,
                        _Float16* __restrict__ mvn) {
  const int lane = threadIdx.x & 31, w = threadIdx.x >> 5;
  const int r = blockIdx.x * 8 + w;
  if (r >= NCLS) return;
  const float inv = 1.f / fmaxf(counts[r], 1.f);
  const float* row = mvsum + (size_t)r * FEAT;
  float m[16]; float ss = 0.f;
#pragma unroll
  for (int q = 0; q < 16; ++q) { float x = row[q * 32 + lane] * inv; m[q] = x; ss += x * x; }
#pragma unroll
  for (int o = 16; o > 0; o >>= 1) ss += __shfl_xor(ss, o, 32);
  const float innorm = rsqrtf(fmaxf(ss, 1e-24f));
  _Float16* orow = mvn + (size_t)r * FEAT;
#pragma unroll
  for (int q = 0; q < 16; ++q) orow[q * 32 + lane] = (_Float16)(m[q] * innorm);
}

// ---------------------------------------------- out1 row L2-normalize -> f16
__global__ void normOutK(const float* __restrict__ out1, _Float16* __restrict__ outn) {
  const int lane = threadIdx.x & 31, w = threadIdx.x >> 5;
  const int r = blockIdx.x * 8 + w;                   // < BATCH
  const float4* src = (const float4*)(out1 + (size_t)r * FEAT);
  float4 v[4]; float ss = 0.f;
#pragma unroll
  for (int q = 0; q < 4; ++q) {
    v[q] = src[q * 32 + lane];
    ss += v[q].x * v[q].x + v[q].y * v[q].y + v[q].z * v[q].z + v[q].w * v[q].w;
  }
#pragma unroll
  for (int o = 16; o > 0; o >>= 1) ss += __shfl_xor(ss, o, 32);
  const float innorm = rsqrtf(fmaxf(ss, 1e-24f));
  h4* dst = (h4*)(outn + (size_t)r * FEAT);
#pragma unroll
  for (int q = 0; q < 4; ++q) {
    h4 hv;
    hv.x = (_Float16)(v[q].x * innorm); hv.y = (_Float16)(v[q].y * innorm);
    hv.z = (_Float16)(v[q].z * innorm); hv.w = (_Float16)(v[q].w * innorm);
    dst[q * 32 + lane] = hv;
  }
}

// ---------------- per-row softmax stats (online), argmax, label-column logit
__global__ void rowStatsK(const float* __restrict__ output,
                          const int* __restrict__ index,
                          const int* __restrict__ slabels,
                          float* __restrict__ rowmax, float* __restrict__ rowsum,
                          int* __restrict__ amax, float* __restrict__ logitc,
                          int* __restrict__ cArr) {
  const int lane = threadIdx.x & 31, w = threadIdx.x >> 5;
  const int r = blockIdx.x * 8 + w;                  // < BATCH
  const float* row = output + (size_t)r * NCLS;
  float m = -INFINITY, s = 0.f, bv = -INFINITY; int bi = NCLS;
  for (int j = lane; j < NCLS; j += 32) {
    const float x = row[j];
    if (x > bv) { bv = x; bi = j; }
    if (x > m) { s = s * __expf(m - x) + 1.f; m = x; }
    else       { s += __expf(x - m); }
  }
#pragma unroll
  for (int o = 16; o > 0; o >>= 1) {
    const float om = __shfl_xor(m, o, 32), os = __shfl_xor(s, o, 32);
    const float nm = fmaxf(m, om);
    s = s * __expf(m - nm) + os * __expf(om - nm);
    m = nm;
    const float obv = __shfl_xor(bv, o, 32); const int obi = __shfl_xor(bi, o, 32);
    if (obv > bv || (obv == bv && obi < bi)) { bv = obv; bi = obi; }
  }
  if (lane == 0) {
    const int cls = slabels[index[r]];
    cArr[r] = cls; logitc[r] = row[cls];
    rowmax[r] = m; rowsum[r] = s; amax[r] = bi;
  }
}

// ----------- batched masked dot products via WMMA: sim[i]=relu(out_n[i].mv_n[c_i])
// One wave per 16 batch rows. A = 16 out_n rows, B = 16 gathered mv_n rows (as
// columns); diagonal of the 16x16 f32 accumulator = the wanted dots.
__global__ void wmmaDotK(const _Float16* __restrict__ outn,
                         const _Float16* __restrict__ mvn,
                         const int* __restrict__ cArr,
                         float* __restrict__ sim) {
  __shared__ _Float16 As[16 * FEAT];
  __shared__ _Float16 Bs[16 * FEAT];
  __shared__ float    Cs[32 * 8];
  const int g = blockIdx.x;                 // row-group
  const int lane = threadIdx.x;             // blockDim == 32
  // Stage 16 A rows + 16 gathered B rows into LDS (uint4 = 8 halves per store).
  for (int r = 0; r < 16; ++r) {
    const uint4* sa = (const uint4*)(outn + (size_t)(g * 16 + r) * FEAT);
    uint4* da = (uint4*)(As + r * FEAT);
    da[lane * 2]     = sa[lane * 2];
    da[lane * 2 + 1] = sa[lane * 2 + 1];
    const int cls = cArr[g * 16 + r];
    const uint4* sb = (const uint4*)(mvn + (size_t)cls * FEAT);
    uint4* db = (uint4*)(Bs + r * FEAT);
    db[lane * 2]     = sb[lane * 2];
    db[lane * 2 + 1] = sb[lane * 2 + 1];
  }
  __syncthreads();
  // Per-lane fragment addressing per CDNA5 ISA layouts (wave32):
  //  A 16x32 f16: lanes 0-15 (M=lane)  hold K{0..7,16..23}; lanes 16-31 K{8..15,24..31}
  //  B 32x16 f16: lanes 0-15 (N=lane)  hold K{0..7,16..23}; lanes 16-31 K{8..15,24..31}
  const int hi = (lane < 16) ? 0 : 1;
  const int rA = lane & 15;
  v8f c = {0.f, 0.f, 0.f, 0.f, 0.f, 0.f, 0.f, 0.f};
#pragma unroll
  for (int ch = 0; ch < FEAT / 32; ++ch) {
    const int baseA = rA * FEAT + ch * 32 + hi * 8;
    const h8 alo = *(const h8*)&As[baseA];
    const h8 ahi = *(const h8*)&As[baseA + 16];
    const int baseB = rA * FEAT + ch * 32 + hi * 8;
    const h8 blo = *(const h8*)&Bs[baseB];
    const h8 bhi = *(const h8*)&Bs[baseB + 16];
    v16h a, b;
#pragma unroll
    for (int k = 0; k < 8; ++k) {
      a[k] = alo[k]; a[8 + k] = ahi[k];
      b[k] = blo[k]; b[8 + k] = bhi[k];
    }
    c = __builtin_amdgcn_wmma_f32_16x16x32_f16(false, a, false, b,
                                               (short)0, c, false, false);
  }
  // Extract diagonal: D(i,i) lives at (VGPR i, lane i) for i<8,
  // (VGPR i-8, lane i+16) for i>=8.
#pragma unroll
  for (int j = 0; j < 8; ++j) Cs[lane * 8 + j] = c[j];
  __syncthreads();
  if (lane < 16) {
    const float d = (lane < 8) ? Cs[lane * 8 + lane]
                               : Cs[(lane + 16) * 8 + (lane - 8)];
    sim[g * 16 + lane] = d > 0.f ? d : 0.f;    // relu == (sim>0)*sim, one-hot mask implicit
  }
}

// ---------------------------------------------------------------- final reduce
__device__ __forceinline__ float blockReduceSum(float v, float* red) {
  const int t = threadIdx.x;
  red[t] = v; __syncthreads();
  for (int s = blockDim.x >> 1; s > 0; s >>= 1) {
    if (t < s) red[t] += red[t + s];
    __syncthreads();
  }
  const float r = red[0]; __syncthreads();
  return r;
}
__device__ __forceinline__ float blockReduceMax(float v, float* red) {
  const int t = threadIdx.x;
  red[t] = v; __syncthreads();
  for (int s = blockDim.x >> 1; s > 0; s >>= 1) {
    if (t < s) red[t] = fmaxf(red[t], red[t + s]);
    __syncthreads();
  }
  const float r = red[0]; __syncthreads();
  return r;
}

__global__ void finalK(const float* __restrict__ sim, const float* __restrict__ rowmax,
                       const float* __restrict__ rowsum, const int* __restrict__ amax,
                       const float* __restrict__ logitc, const int* __restrict__ cArr,
                       const int* __restrict__ index, const float* __restrict__ u,
                       const float* __restrict__ tacp, float* __restrict__ out) {
  __shared__ float red[1024];
  const int t = threadIdx.x;
  const float tac = tacp[0];
  // Pass A: batch maxes for the two log-sum-exps (over logit_c and -log u).
  float m1 = -INFINITY, m2 = -INFINITY;
  for (int i = t; i < BATCH; i += blockDim.x) {
    m1 = fmaxf(m1, logitc[i]);
    m2 = fmaxf(m2, -logf(u[index[i]]));
  }
  const float M1 = blockReduceMax(m1, red);
  const float M2 = blockReduceMax(m2, red);
  // Pass B: partition sums.
  float z1 = 0.f, z2 = 0.f;
  for (int i = t; i < BATCH; i += blockDim.x) {
    z1 += __expf(logitc[i] - M1);
    z2 += __expf(-logf(u[index[i]]) - M2);
  }
  const float Z1 = blockReduceSum(z1, red);
  const float Z2 = blockReduceSum(z2, red);
  const float lZ1 = logf(Z1), lZ2 = logf(Z2);
  // Pass C: all three loss terms.
  float l1 = 0.f, l2 = 0.f, kl = 0.f;
  for (int i = t; i < BATCH; i += blockDim.x) {
    const float uv = u[index[i]];
    const float lc = logitc[i];
    // CE term at the label column of the clipped softmax
    const float p  = __expf(lc - rowmax[i]) / rowsum[i];
    float pc = p + tac * uv;
    pc = fminf(fmaxf(pc, EPSV), 1.0f);
    l1 += -sim[i] * logf(pc);
    // MSE of (one_hot(argmax) + u_b - label): closed form
    const float um1 = uv - 1.f;
    l2 += (amax[i] == cArr[i]) ? uv * uv : (1.f + um1 * um1);
    // KL(t || softmax(logits_b)) pieces
    const float nu = -logf(uv);
    const float ti = __expf(nu - M2) / Z2;
    const float logti = (nu - M2) - lZ2;
    const float logpi = (lc - M1) - lZ1;
    kl += ti * (logti - logpi);
  }
  const float L1 = blockReduceSum(l1, red);
  const float L2 = blockReduceSum(l2, red);
  const float KL = blockReduceSum(kl, red);
  if (t == 0)
    out[0] = (L1 + L2) / (float)BATCH + (1.f - tac) * KL / (float)BATCH;
}

// ------------------------------------------------------------------- launcher
extern "C" void kernel_launch(void* const* d_in, const int* in_sizes, int n_in,
                              void* d_out, int out_size, void* d_ws, size_t ws_size,
                              hipStream_t stream) {
  const int*   index  = (const int*)  d_in[0];
  const float* output = (const float*)d_in[1];
  // d_in[2] (label) is one_hot(sample_labels[index]) -> derived from d_in[0]/d_in[6]
  const float* out1   = (const float*)d_in[3];
  const float* u      = (const float*)d_in[4];
  const float* prevS  = (const float*)d_in[5];
  const int*   slab   = (const int*)  d_in[6];
  const float* tac    = (const float*)d_in[7];

  char* ws = (char*)d_ws;
  float*    mvsum  = (float*)   (ws + 0);          // 1000*512*4  = 2,048,000
  float*    counts = (float*)   (ws + 2048000);    // 4,096 (padded)
  _Float16* mvn    = (_Float16*)(ws + 2052096);    // 1000*512*2  = 1,024,000
  _Float16* outn   = (_Float16*)(ws + 3076096);    // 16384*512*2 = 16,777,216
  int*      cArr   = (int*)     (ws + 19853312);   // 65,536
  float*    rowmax = (float*)   (ws + 19918848);   // 65,536
  float*    rowsum = (float*)   (ws + 19984384);   // 65,536
  int*      amaxA  = (int*)     (ws + 20049920);   // 65,536
  float*    logitc = (float*)   (ws + 20115456);   // 65,536
  float*    sim    = (float*)   (ws + 20180992);   // 65,536  (total ~20.2 MB)

  zeroK  <<<512, 256, 0, stream>>>(mvsum, (2048000 + 4096) / 4);
  countK <<<128, 256, 0, stream>>>(slab, counts);
  segsumK<<<SEG_FB * SEG_EC, 256, NCLS * 64 * sizeof(float), stream>>>(prevS, slab, mvsum);
  normMvK<<<(NCLS + 7) / 8, 256, 0, stream>>>(mvsum, counts, mvn);
  rowStatsK<<<BATCH / 8, 256, 0, stream>>>(output, index, slab,
                                           rowmax, rowsum, amaxA, logitc, cArr);
  normOutK<<<BATCH / 8, 256, 0, stream>>>(out1, outn);
  wmmaDotK<<<BATCH / 16, 32, 0, stream>>>(outn, mvn, cArr, sim);
  finalK<<<1, 1024, 0, stream>>>(sim, rowmax, rowsum, amaxA, logitc, cArr,
                                 index, u, tac, (float*)d_out);
}
